// LyraContact_83906481094705
// MI455X (gfx1250) — matrix operand order repaired and checked by reading, not compiled, not run on hardware
//
#include <hip/hip_runtime.h>
#include <hip/hip_bf16.h>
#include <math.h>

#define BB 8
#define LL 2048
#define DD 256
#define M_ROWS (BB * LL)
#define OUTW (LL + 6)   // 3 + 2 + L + 1 = 2054

typedef __attribute__((ext_vector_type(16))) __bf16 v16bf;
typedef __attribute__((ext_vector_type(8)))  float  v8f;

__device__ __forceinline__ float gelu_exact(float v) {
    return 0.5f * v * (1.0f + erff(v * 0.70710678118654752f));
}

// --- WMMA A-fragment: 16x32 bf16, row r = lane&15, half = lane>>4 ---
// VGPR j holds K pair: k = (j&4 ? 16 : 0) + half*8 + (j&3)*2
__device__ __forceinline__ v16bf load_afrag(const float* __restrict__ Arow,
                                            int k0, int half) {
    v16bf a;
#pragma unroll
    for (int j = 0; j < 8; ++j) {
        int kk = k0 + ((j & 4) << 2) + half * 8 + ((j & 3) << 1);
        a[2 * j]     = (__bf16)Arow[kk];
        a[2 * j + 1] = (__bf16)Arow[kk + 1];
    }
    return a;
}

// ---------------- Encoder: h = x @ enc_w + enc_b  (DIN=4) ----------------
__global__ void k_encoder(const float* __restrict__ x, const float* __restrict__ w,
                          const float* __restrict__ b, float* __restrict__ h) {
    int idx = blockIdx.x * blockDim.x + threadIdx.x;   // B*L*D
    int d = idx & (DD - 1);
    int bl = idx >> 8;
    float s = b[d];
#pragma unroll
    for (int i = 0; i < 4; ++i) s += x[(size_t)bl * 4 + i] * w[i * DD + d];
    h[idx] = s;
}

// ---------------- Generic bf16-WMMA GEMM: C = A[M,K] @ W + bias ----------
// Fully compile-time specialized: K, N, TRANSB, GELU are template params so
// all W addresses are base + constant immediate offsets (no runtime selects).
// TRANSB=0: W is (K,N) row-major.  TRANSB=1: W is (N,K) row-major.
// One wave computes a 16x64 macro-tile (4 WMMA accumulators, shared A frag).
// Grid: x = N/64, y = M/16.
template <int K, int N, bool TRANSB, bool GELU>
__global__ void __launch_bounds__(32) k_gemm(
        const float* __restrict__ A, const float* __restrict__ W,
        const float* __restrict__ bias, float* __restrict__ C) {
    int lane = threadIdx.x & 31;
    int half = lane >> 4;
    int r    = lane & 15;
    int n0   = blockIdx.x << 6;
    const float* Arow = A + (size_t)((blockIdx.y << 4) + r) * K;

    v8f acc[4] = {{}, {}, {}, {}};
#pragma unroll 2
    for (int k0 = 0; k0 < K; k0 += 32) {
        __builtin_prefetch(Arow + k0 + 64, 0, 0);  // global_prefetch_b8
        v16bf a = load_afrag(Arow, k0, half);
        int kb = k0;                               // B frag: lane&15 = N col, elem e = K (table order)
        const float* Wk = TRANSB ? (W + (size_t)(n0 + r) * K + kb)
                                 : (W + (size_t)kb * N + n0 + r);
#pragma unroll
        for (int t = 0; t < 4; ++t) {
            v16bf bf;
#pragma unroll
            for (int e = 0; e < 16; ++e) {
                int ke = ((e & 8) << 1) + half * 8 + (e & 7);
                float wv = TRANSB ? Wk[(size_t)(t << 4) * K + ke]
                                  : Wk[(size_t)ke * N + (t << 4)];
                bf[e] = (__bf16)wv;
            }
            acc[t] = __builtin_amdgcn_wmma_f32_16x16x32_bf16(
                         false, a, false, bf, (short)0, acc[t], false, false);
        }
    }
    int row = (blockIdx.y << 4) + half * 8;
#pragma unroll
    for (int t = 0; t < 4; ++t) {
        int col = n0 + (t << 4) + r;
        float bv = bias[col];
#pragma unroll
        for (int i = 0; i < 8; ++i) {
            float v = acc[t][i] + bv;
            if (GELU) v = gelu_exact(v);
            C[(size_t)(row + i) * N + col] = v;
        }
    }
}

// ---------------- Row-wise RMSNorm over width E ----------------
__global__ void k_rmsnorm(const float* __restrict__ X, const float* __restrict__ w,
                          float* __restrict__ Y, int E, float eps) {
    int row = blockIdx.x;
    const float* x = X + (size_t)row * E;
    float* y = Y + (size_t)row * E;
    float ss = 0.f;
    for (int e = threadIdx.x; e < E; e += blockDim.x) { float v = x[e]; ss += v * v; }
#pragma unroll
    for (int m = 16; m >= 1; m >>= 1) ss += __shfl_xor(ss, m, 32);
    __shared__ float red[8];
    int wid = threadIdx.x >> 5, lid = threadIdx.x & 31;
    if (lid == 0) red[wid] = ss;
    __syncthreads();
    if (wid == 0) {
        float t = (lid < (int)(blockDim.x >> 5)) ? red[lid] : 0.f;
#pragma unroll
        for (int m = 16; m >= 1; m >>= 1) t += __shfl_xor(t, m, 32);
        if (lid == 0) red[0] = t;
    }
    __syncthreads();
    float r = rsqrtf(red[0] / (float)E + eps);
    for (int e = threadIdx.x; e < E; e += blockDim.x) y[e] = x[e] * r * w[e];
}

// ---------------- Depthwise conv3 + gate: g = v * (conv(x) + cb) ----------
__global__ void k_conv_gate(const float* __restrict__ xv, const float* __restrict__ cw,
                            const float* __restrict__ cb, float* __restrict__ g) {
    int idx = blockIdx.x * blockDim.x + threadIdx.x;   // B*L*D
    int e  = idx & (DD - 1);
    int bl = idx >> 8;
    int l  = bl & (LL - 1);
    const float* xr = xv + (size_t)bl * (2 * DD);
    float x0 = xr[e];
    float xm = (l > 0)      ? xv[(size_t)(bl - 1) * (2 * DD) + e] : 0.f;
    float xp = (l < LL - 1) ? xv[(size_t)(bl + 1) * (2 * DD) + e] : 0.f;
    float conv = cw[e * 3 + 0] * xm + cw[e * 3 + 1] * x0 + cw[e * 3 + 2] * xp + cb[e];
    g[idx] = xr[DD + e] * conv;
}

// ---------------- S4D diagonal recurrence (exact FFT-conv equivalent) ----
// One wave32 per (b,channel); lane n = complex mode n (N/2 = 32 modes).
__global__ void __launch_bounds__(32) k_s4_scan(
        const float* __restrict__ z, const float* __restrict__ Dp,
        const float* __restrict__ Cri, const float* __restrict__ ldt,
        const float* __restrict__ lAr, const float* __restrict__ Aim,
        float* __restrict__ y) {
    int bh = blockIdx.x;
    int b = bh >> 8, h = bh & (DD - 1);
    int n = threadIdx.x & 31;
    float dt  = expf(ldt[h]);
    float Ar  = -expf(lAr[h * 32 + n]);
    float Ai  = Aim[h * 32 + n];
    float dAr = Ar * dt, dAi = Ai * dt;
    float er  = expf(dAr);
    float lr  = er * cosf(dAi), li = er * sinf(dAi);      // lambda = exp(dt*A)
    float c0r = Cri[(size_t)(h * 32 + n) * 2 + 0];
    float c0i = Cri[(size_t)(h * 32 + n) * 2 + 1];
    float mr = lr - 1.f, mi = li;                          // exp(dtA) - 1
    float tr = c0r * mr - c0i * mi, ti = c0r * mi + c0i * mr;
    float den = Ar * Ar + Ai * Ai;
    float cr = (tr * Ar + ti * Ai) / den;                  // Cm = C*(e^{dtA}-1)/A
    float ci = (ti * Ar - tr * Ai) / den;
    float dp = Dp[h];
    float sr = 0.f, si = 0.f;
    const float* zin = z + (size_t)b * LL * DD + h;
    float*       yo  = y + (size_t)b * LL * DD + h;
    for (int l = 0; l < LL; ++l) {
        float u  = zin[(size_t)l * DD];
        float ns = lr * sr - li * si + u;
        float nc = li * sr + lr * si;
        sr = ns; si = nc;
        float c = cr * sr - ci * si;
#pragma unroll
        for (int m = 16; m >= 1; m >>= 1) c += __shfl_xor(c, m, 32);
        if (n == 0) {
            float v = 2.f * c + u * dp;
            yo[(size_t)l * DD] = gelu_exact(v);
        }
    }
}

// ---------------- GLU + residual: h += a * sigmoid(g) ----------------
__global__ void k_glu_res(const float* __restrict__ z, float* __restrict__ h) {
    int idx = blockIdx.x * blockDim.x + threadIdx.x;   // B*L*D
    int o  = idx & (DD - 1);
    int bl = idx >> 8;
    float a = z[(size_t)bl * (2 * DD) + o];
    float g = z[(size_t)bl * (2 * DD) + DD + o];
    h[idx] += a / (1.f + expf(-g));
}

// ---------------- Scores: out[b,q,5+k] = (Q_b @ K_b^T)/16, WMMA ----------
// One wave computes a 16x64 macro-tile of the LxL score matrix.
// Grid: x = L/64, y = L/16, z = B.
__global__ void __launch_bounds__(32) k_scores(
        const float* __restrict__ Q, const float* __restrict__ Kc,
        float* __restrict__ out) {
    int b = blockIdx.z;
    int lane = threadIdx.x & 31;
    int half = lane >> 4;
    int r    = lane & 15;
    int n0   = blockIdx.x << 6;
    const float* Arow = Q  + (size_t)(b * LL + (blockIdx.y << 4) + r) * DD;
    const float* Kb   = Kc + (size_t)b * LL * DD;
    v8f acc[4] = {{}, {}, {}, {}};
#pragma unroll 2
    for (int k0 = 0; k0 < DD; k0 += 32) {
        __builtin_prefetch(Arow + k0 + 64, 0, 0);
        v16bf a = load_afrag(Arow, k0, half);
        int kb = k0;
        const float* Kk = Kb + (size_t)(n0 + r) * DD + kb;   // K^T: constant stride DD
#pragma unroll
        for (int t = 0; t < 4; ++t) {
            v16bf bf;
#pragma unroll
            for (int e = 0; e < 16; ++e)
                bf[e] = (__bf16)Kk[(size_t)(t << 4) * DD + ((e & 8) << 1) + half * 8 + (e & 7)];
            acc[t] = __builtin_amdgcn_wmma_f32_16x16x32_bf16(
                         false, a, false, bf, (short)0, acc[t], false, false);
        }
    }
    int row = (blockIdx.y << 4) + half * 8;
    float* ob = out + (size_t)b * LL * OUTW;
#pragma unroll
    for (int t = 0; t < 4; ++t) {
        int col = 5 + n0 + (t << 4) + r;
#pragma unroll
        for (int i = 0; i < 8; ++i)
            ob[(size_t)(row + i) * OUTW + col] = acc[t][i] * 0.0625f; // 1/sqrt(256)
    }
}

// ---------------- Tiny head finishers + unpaired bias ----------------
__global__ void k_final(const float* __restrict__ db1, const float* __restrict__ dw2,
                        const float* __restrict__ db2, const float* __restrict__ bin1,
                        const float* __restrict__ bw2, const float* __restrict__ bb2,
                        const float* __restrict__ unp, float* __restrict__ out) {
    int idx = blockIdx.x * blockDim.x + threadIdx.x;   // B*L
    if (idx >= M_ROWS) return;
    float* o = out + (size_t)idx * OUTW;
    const float* d1 = db1 + (size_t)idx * 128;
#pragma unroll
    for (int j = 0; j < 3; ++j) {
        float s = db2[j];
        for (int t = 0; t < 128; ++t) s += d1[t] * dw2[t * 3 + j];
        o[j] = s;
    }
    const float* b1 = bin1 + (size_t)idx * 64;
#pragma unroll
    for (int j = 0; j < 2; ++j) {
        float s = bb2[j];
        for (int t = 0; t < 64; ++t) s += b1[t] * bw2[t * 2 + j];
        o[3 + j] = s;
    }
    o[5 + LL] = unp[0];
}

extern "C" void kernel_launch(void* const* d_in, const int* in_sizes, int n_in,
                              void* d_out, int out_size, void* d_ws, size_t ws_size,
                              hipStream_t stream) {
    const float* x            = (const float*)d_in[0];
    const float* enc_w        = (const float*)d_in[1];
    const float* enc_b        = (const float*)d_in[2];
    const float* pgc_in_w     = (const float*)d_in[3];
    const float* pgc_in_b     = (const float*)d_in[4];
    const float* pgc_innorm_w = (const float*)d_in[5];
    const float* pgc_conv_w   = (const float*)d_in[6];
    const float* pgc_conv_b   = (const float*)d_in[7];
    const float* pgc_norm_w   = (const float*)d_in[8];
    const float* pgc_out_w    = (const float*)d_in[9];
    const float* pgc_out_b    = (const float*)d_in[10];
    const float* s4_norm_w    = (const float*)d_in[11];
    const float* s4_D         = (const float*)d_in[12];
    const float* s4_C         = (const float*)d_in[13];
    const float* s4_log_dt    = (const float*)d_in[14];
    const float* s4_log_A_r   = (const float*)d_in[15];
    const float* s4_A_imag    = (const float*)d_in[16];
    const float* s4_out_w     = (const float*)d_in[17];
    const float* s4_out_b     = (const float*)d_in[18];
    const float* db_w1        = (const float*)d_in[19];
    const float* db_b1        = (const float*)d_in[20];
    const float* db_w2        = (const float*)d_in[21];
    const float* db_b2        = (const float*)d_in[22];
    const float* bin_w1       = (const float*)d_in[23];
    const float* bin_b1       = (const float*)d_in[24];
    const float* bin_w2       = (const float*)d_in[25];
    const float* bin_b2       = (const float*)d_in[26];
    const float* q_w          = (const float*)d_in[27];
    const float* q_b          = (const float*)d_in[28];
    const float* k_w          = (const float*)d_in[29];
    const float* k_b          = (const float*)d_in[30];
    const float* unpaired     = (const float*)d_in[31];
    float* out = (float*)d_out;

    float* ws    = (float*)d_ws;
    float* bufH  = ws;                                   // (M,256)
    float* bufXV = bufH  + (size_t)M_ROWS * DD;          // (M,512)
    float* bufG  = bufXV + (size_t)M_ROWS * 2 * DD;      // (M,256) also db1 (M,128)
    float* bufY  = bufG  + (size_t)M_ROWS * DD;          // (M,256) also bin1 (M,64)
    float* bufZ  = bufY  + (size_t)M_ROWS * DD;          // (M,512)

    const float EPS_T = 1.1920929e-07f;
    const float EPS_C = 1e-8f;

    // Encoder
    k_encoder<<<(M_ROWS * DD) / 256, 256, 0, stream>>>(x, enc_w, enc_b, bufH);

    // PGC stack
    for (int i = 0; i < 4; ++i) {
        k_gemm<256, 512, false, false><<<dim3(512 / 64, M_ROWS / 16), 32, 0, stream>>>(
            bufH, pgc_in_w + (size_t)i * DD * 2 * DD, pgc_in_b + (size_t)i * 2 * DD, bufXV);
        k_rmsnorm<<<M_ROWS, 256, 0, stream>>>(bufXV, pgc_innorm_w + (size_t)i * 2 * DD,
                                              bufXV, 2 * DD, EPS_T);
        k_conv_gate<<<(M_ROWS * DD) / 256, 256, 0, stream>>>(
            bufXV, pgc_conv_w + (size_t)i * DD * 3, pgc_conv_b + (size_t)i * DD, bufG);
        k_rmsnorm<<<M_ROWS, 256, 0, stream>>>(bufG, pgc_norm_w + (size_t)i * DD,
                                              bufG, DD, EPS_T);
        k_gemm<256, 256, false, false><<<dim3(256 / 64, M_ROWS / 16), 32, 0, stream>>>(
            bufG, pgc_out_w + (size_t)i * DD * DD, pgc_out_b + (size_t)i * DD, bufH);
    }

    // S4D residual stack (prenorm; recurrence replaces FFT conv exactly)
    for (int i = 0; i < 4; ++i) {
        k_rmsnorm<<<M_ROWS, 256, 0, stream>>>(bufH, s4_norm_w + (size_t)i * DD,
                                              bufY, DD, EPS_C);
        k_s4_scan<<<BB * DD, 32, 0, stream>>>(
            bufY, s4_D + (size_t)i * DD, s4_C + (size_t)i * DD * 64,
            s4_log_dt + (size_t)i * DD, s4_log_A_r + (size_t)i * DD * 32,
            s4_A_imag + (size_t)i * DD * 32, bufY);
        k_gemm<256, 512, true, false><<<dim3(512 / 64, M_ROWS / 16), 32, 0, stream>>>(
            bufY, s4_out_w + (size_t)i * 2 * DD * DD, s4_out_b + (size_t)i * 2 * DD, bufZ);
        k_glu_res<<<(M_ROWS * DD) / 256, 256, 0, stream>>>(bufZ, bufH);
    }

    // Heads
    float* bufQ = bufXV;
    float* bufK = bufXV + (size_t)M_ROWS * DD;
    k_gemm<256, 256, false, false><<<dim3(256 / 64, M_ROWS / 16), 32, 0, stream>>>(
        bufH, q_w, q_b, bufQ);
    k_gemm<256, 256, false, false><<<dim3(256 / 64, M_ROWS / 16), 32, 0, stream>>>(
        bufH, k_w, k_b, bufK);
    k_gemm<256, 128, false, true><<<dim3(128 / 64, M_ROWS / 16), 32, 0, stream>>>(
        bufH, db_w1, db_b1, bufG);
    k_gemm<256, 64, false, true><<<dim3(64 / 64, M_ROWS / 16), 32, 0, stream>>>(
        bufH, bin_w1, bin_b1, bufY);

    k_scores<<<dim3(LL / 64, LL / 16, BB), 32, 0, stream>>>(bufQ, bufK, out);
    k_final<<<(M_ROWS + 255) / 256, 256, 0, stream>>>(
        bufG, db_w2, db_b2, bufY, bin_w2, bin_b2, unpaired, out);
}
